// SudokuRRN_30236569764521
// MI455X (gfx1250) — matrix operand order repaired and checked by reading, not compiled, not run on hardware
//
#include <hip/hip_runtime.h>
#include <hip/hip_bf16.h>

typedef __attribute__((ext_vector_type(16))) __bf16 v16bf;
typedef __attribute__((ext_vector_type(8)))  __bf16 v8bf;
typedef __attribute__((ext_vector_type(8)))  float  v8f;

#define HD   128
#define NN   81
#define BB   512
#define EDG  1620
#define NSTEPS 16
#define LN_EPS 1e-5f

#define GRP    3           // dst nodes per message block (edge list is dst-major, 20 edges/dst)
#define GEDGE  (GRP * 20)  // 60 valid edge rows per block (padded to 64 = 4 wave tiles)
#define GROUPS (NN / GRP)  // 27 groups per batch

// ---------------------------------------------------------------------------
// Pack an fp32 weight matrix W[K][128] into bf16 "B-fragment-major" layout so
// that each lane's WMMA B operand is one contiguous 32-byte load:
//   out[((kc*8 + nt)*32 + lane)*16 + j] = W[kc*32 + (j>>3)*16 + (lane>>4)*8 + (j&7)][nt*16 + (lane&15)]
// ---------------------------------------------------------------------------
__global__ void pack_w_kernel(const float* __restrict__ W, __bf16* __restrict__ out, int K) {
  int idx = blockIdx.x * blockDim.x + threadIdx.x;
  if (idx >= K * HD) return;
  int j    = idx & 15;
  int lane = (idx >> 4) & 31;
  int nt   = (idx >> 9) & 7;
  int kc   = idx >> 12;
  int k = kc * 32 + (j >> 3) * 16 + (lane >> 4) * 8 + (j & 7);
  int n = nt * 16 + (lane & 15);
  out[idx] = (__bf16)W[k * HD + n];
}

// ---------------------------------------------------------------------------
// x_embed = x @ W_in + b_in + pos ;  h = x_embed
// ---------------------------------------------------------------------------
__global__ void embed_kernel(const float* __restrict__ x, const float* __restrict__ W_in,
                             const float* __restrict__ b_in, const float* __restrict__ pos,
                             float* __restrict__ xemb, float* __restrict__ h) {
  int idx = blockIdx.x * blockDim.x + threadIdx.x;
  if (idx >= BB * NN * HD) return;
  int c   = idx & (HD - 1);
  int row = idx >> 7;           // b*81+n
  int n   = row % NN;
  float acc = b_in[c] + pos[n * HD + c];
  const float* xp = x + row * 10;
  #pragma unroll
  for (int k = 0; k < 10; ++k) acc += xp[k] * W_in[k * HD + c];
  xemb[idx] = acc;
  h[idx]    = acc;
}

// ---------------------------------------------------------------------------
// One MLP layer for a 16-row tile held in LDS (bf16, row-major, stride K).
// A-fragment assembly follows the CDNA5 16-bit A layout:
//   lane l: M = l&15, Kbase = (l>>4)*8 ; j<8 -> Kbase+j, j>=8 -> Kbase+16+(j&7)
// B fragments are contiguous 32B loads from the packed weights.
// Epilogue functor receives (m, n, acc+bias).
// ---------------------------------------------------------------------------
template <int K, typename Epi>
__device__ __forceinline__ void mlp_layer(const __bf16* __restrict__ inBuf,
                                          const __bf16* __restrict__ wPacked,
                                          const float* __restrict__ bias,
                                          int lane, Epi epi) {
  constexpr int NKC = K / 32;
  v8f acc[8];
  #pragma unroll
  for (int nt = 0; nt < 8; ++nt) acc[nt] = v8f{0.f, 0.f, 0.f, 0.f, 0.f, 0.f, 0.f, 0.f};
  const int m  = lane & 15;
  const int kb = (lane >> 4) * 8;
  #pragma unroll
  for (int kc = 0; kc < NKC; ++kc) {
    const __bf16* ip = inBuf + m * K + kc * 32 + kb;
    v8bf lo = *(const v8bf*)(ip);
    v8bf hi = *(const v8bf*)(ip + 16);
    v16bf a;
    #pragma unroll
    for (int j = 0; j < 8; ++j) { a[j] = lo[j]; a[j + 8] = hi[j]; }
    #pragma unroll
    for (int nt = 0; nt < 8; ++nt) {
      v16bf b = *(const v16bf*)(wPacked + ((kc * 8 + nt) * 32 + lane) * 16);
      acc[nt] = __builtin_amdgcn_wmma_f32_16x16x32_bf16(
          /*neg_a=*/false, a, /*neg_b=*/false, b,
          /*c_mod=*/(short)0, acc[nt], /*reuse_a=*/false, /*reuse_b=*/false);
    }
  }
  #pragma unroll
  for (int nt = 0; nt < 8; ++nt) {
    const int n = nt * 16 + (lane & 15);
    const float bv = bias[n];
    #pragma unroll
    for (int r = 0; r < 8; ++r) {
      const int mm = r + (lane >> 4) * 8;   // C/D layout: lanes 16-31 hold M+8
      epi(mm, n, acc[nt][r] + bv);
    }
  }
}

// ---------------------------------------------------------------------------
// Edge message MLP + dst-grouped aggregation.
// One block = one batch element x 3 dst nodes (60 edges, padded to 4x16 rows).
// Messages are reduced into per-block LDS (ds_add_f32), then written to agg
// with plain stores: no global atomics, no agg memset needed.
// ---------------------------------------------------------------------------
__global__ void msg_kernel(const float* __restrict__ h,
                           const int* __restrict__ srcIdx, const int* __restrict__ dstIdx,
                           const __bf16* __restrict__ w0, const float* __restrict__ b0,
                           const __bf16* __restrict__ w1, const float* __restrict__ b1,
                           const __bf16* __restrict__ w2, const float* __restrict__ b2,
                           const __bf16* __restrict__ w3, const float* __restrict__ b3,
                           float* __restrict__ agg) {
  extern __shared__ __bf16 smem[];
  const int wave = threadIdx.x >> 5;
  const int lane = threadIdx.x & 31;
  __bf16* inBuf  = smem + wave * (16 * 256 + 16 * 128);
  __bf16* outBuf = inBuf + 16 * 256;
  float*  aggBuf = (float*)(smem + 4 * (16 * 256 + 16 * 128));  // GRP x 128 f32

  const int b     = blockIdx.x / GROUPS;
  const int g     = blockIdx.x - b * GROUPS;
  const int eBase = g * GEDGE;            // first edge (within batch) for this block
  const int rlb   = wave * 16;            // this wave's local row base

  for (int i = threadIdx.x; i < GRP * HD; i += blockDim.x) aggBuf[i] = 0.f;

  // Stage concat(h[src], h[dst]) as bf16, 16 rows x 256 per wave
  for (int i = lane; i < 16 * 256; i += 32) {
    int r = i >> 8, c = i & 255;
    int rl = rlb + r;
    float v = 0.f;
    if (rl < GEDGE) {
      int e = eBase + rl;
      int node = (c < HD) ? srcIdx[e] : dstIdx[e];
      v = h[(b * NN + node) * HD + (c & (HD - 1))];
    }
    inBuf[r * 256 + c] = (__bf16)v;
  }
  __syncthreads();

  mlp_layer<256>(inBuf, w0, b0, lane, [&](int mm, int n, float v) {
    outBuf[mm * HD + n] = (__bf16)(v > 0.f ? v : 0.f);
  });
  __syncthreads();
  mlp_layer<128>(outBuf, w1, b1, lane, [&](int mm, int n, float v) {
    inBuf[mm * HD + n] = (__bf16)(v > 0.f ? v : 0.f);
  });
  __syncthreads();
  mlp_layer<128>(inBuf, w2, b2, lane, [&](int mm, int n, float v) {
    outBuf[mm * HD + n] = (__bf16)(v > 0.f ? v : 0.f);
  });
  __syncthreads();
  // Final layer: reduce 20 messages per dst into LDS (edge list is dst-major)
  mlp_layer<128>(outBuf, w3, b3, lane, [&](int mm, int n, float v) {
    int rl = rlb + mm;
    if (rl < GEDGE) atomicAdd(&aggBuf[(rl / 20) * HD + n], v);
  });
  __syncthreads();

  // Plain coalesced stores: each (b, dst) owned by exactly one block
  for (int i = threadIdx.x; i < GRP * HD; i += blockDim.x) {
    int dl = i >> 7, n = i & (HD - 1);
    agg[(b * NN + g * GRP + dl) * HD + n] = aggBuf[i];
  }
}

// ---------------------------------------------------------------------------
// Node MLP + residual + LayerNorm.  One wave = 16 node-rows x 128 cols.
// ---------------------------------------------------------------------------
__global__ void node_kernel(float* __restrict__ h,
                            const float* __restrict__ xemb, const float* __restrict__ agg,
                            const __bf16* __restrict__ w0, const float* __restrict__ b0,
                            const __bf16* __restrict__ w1, const float* __restrict__ b1,
                            const __bf16* __restrict__ w2, const float* __restrict__ b2,
                            const __bf16* __restrict__ w3, const float* __restrict__ b3,
                            const float* __restrict__ ln_g, const float* __restrict__ ln_b) {
  extern __shared__ __bf16 smem[];
  const int wave = threadIdx.x >> 5;
  const int lane = threadIdx.x & 31;
  __bf16* inBuf  = smem + wave * (16 * 384 + 16 * 128);
  __bf16* midBuf = inBuf + 16 * 384;
  float*  fBuf   = (float*)inBuf;   // reuse input region (only needed by layer 0)
  const int rowBase = (blockIdx.x * 4 + wave) * 16;  // B*N divisible by 64

  // Stage concat(h, x_embed, agg) as bf16, 16 rows x 384
  for (int r = 0; r < 16; ++r) {
    int row = rowBase + r;
    for (int c = lane; c < 384; c += 32) {
      float v;
      if (c < HD)            v = h[row * HD + c];
      else if (c < 2 * HD)   v = xemb[row * HD + (c - HD)];
      else                   v = agg[row * HD + (c - 2 * HD)];
      inBuf[r * 384 + c] = (__bf16)v;
    }
  }
  __syncthreads();

  mlp_layer<384>(inBuf, w0, b0, lane, [&](int mm, int n, float v) {
    midBuf[mm * HD + n] = (__bf16)(v > 0.f ? v : 0.f);
  });
  __syncthreads();
  mlp_layer<128>(midBuf, w1, b1, lane, [&](int mm, int n, float v) {
    midBuf[mm * HD + n] = (__bf16)(v > 0.f ? v : 0.f);   // in-place: wave reads precede stores
  });
  __syncthreads();
  mlp_layer<128>(midBuf, w2, b2, lane, [&](int mm, int n, float v) {
    midBuf[mm * HD + n] = (__bf16)(v > 0.f ? v : 0.f);
  });
  __syncthreads();
  mlp_layer<128>(midBuf, w3, b3, lane, [&](int mm, int n, float v) {
    int row = rowBase + mm;
    fBuf[mm * HD + n] = h[row * HD + n] + v;             // residual, f32
  });
  __syncthreads();

  // LayerNorm per row; lanes 0..15 each own one row of this wave's tile
  if (lane < 16) {
    int row = rowBase + lane;
    const float* rp = fBuf + lane * HD;
    float mu = 0.f;
    for (int c = 0; c < HD; ++c) mu += rp[c];
    mu *= (1.f / HD);
    float var = 0.f;
    for (int c = 0; c < HD; ++c) { float d = rp[c] - mu; var += d * d; }
    var *= (1.f / HD);
    float inv = rsqrtf(var + LN_EPS);
    for (int c = 0; c < HD; ++c)
      h[row * HD + c] = (rp[c] - mu) * inv * ln_g[c] + ln_b[c];
  }
}

// ---------------------------------------------------------------------------
// out = h @ W_out + b_out   (K=128, N=9 — tiny, plain VALU)
// ---------------------------------------------------------------------------
__global__ void out_kernel(const float* __restrict__ h, const float* __restrict__ Wout,
                           const float* __restrict__ bout, float* __restrict__ out) {
  int idx = blockIdx.x * blockDim.x + threadIdx.x;
  if (idx >= BB * NN * 9) return;
  int c   = idx % 9;
  int row = idx / 9;
  float acc = bout[c];
  const float* hp = h + row * HD;
  #pragma unroll 8
  for (int k = 0; k < HD; ++k) acc += hp[k] * Wout[k * 9 + c];
  out[idx] = acc;
}

extern "C" void kernel_launch(void* const* d_in, const int* in_sizes, int n_in,
                              void* d_out, int out_size, void* d_ws, size_t ws_size,
                              hipStream_t stream) {
  (void)in_sizes; (void)n_in; (void)out_size; (void)ws_size;
  const float* x    = (const float*)d_in[0];
  const int*   src  = (const int*)d_in[1];
  const int*   dst  = (const int*)d_in[2];
  const float* W_in = (const float*)d_in[3];
  const float* b_in = (const float*)d_in[4];
  const float* pos  = (const float*)d_in[5];
  const float* mW0  = (const float*)d_in[6];  const float* mb0 = (const float*)d_in[7];
  const float* mW1  = (const float*)d_in[8];  const float* mb1 = (const float*)d_in[9];
  const float* mW2  = (const float*)d_in[10]; const float* mb2 = (const float*)d_in[11];
  const float* mW3  = (const float*)d_in[12]; const float* mb3 = (const float*)d_in[13];
  const float* nW0  = (const float*)d_in[14]; const float* nb0 = (const float*)d_in[15];
  const float* nW1  = (const float*)d_in[16]; const float* nb1 = (const float*)d_in[17];
  const float* nW2  = (const float*)d_in[18]; const float* nb2 = (const float*)d_in[19];
  const float* nW3  = (const float*)d_in[20]; const float* nb3 = (const float*)d_in[21];
  const float* ln_g = (const float*)d_in[22]; const float* ln_b = (const float*)d_in[23];
  const float* Wout = (const float*)d_in[24]; const float* bout = (const float*)d_in[25];

  char* ws = (char*)d_ws;
  size_t off = 0;
  auto carve = [&](size_t bytes) -> void* {
    void* p = ws + off;
    off = (off + bytes + 255) & ~(size_t)255;
    return p;
  };
  const size_t BNH = (size_t)BB * NN * HD;
  float* h    = (float*)carve(BNH * 4);
  float* xemb = (float*)carve(BNH * 4);
  float* agg  = (float*)carve(BNH * 4);
  __bf16* mW0p = (__bf16*)carve((size_t)256 * 128 * 2);
  __bf16* mW1p = (__bf16*)carve((size_t)128 * 128 * 2);
  __bf16* mW2p = (__bf16*)carve((size_t)128 * 128 * 2);
  __bf16* mW3p = (__bf16*)carve((size_t)128 * 128 * 2);
  __bf16* nW0p = (__bf16*)carve((size_t)384 * 128 * 2);
  __bf16* nW1p = (__bf16*)carve((size_t)128 * 128 * 2);
  __bf16* nW2p = (__bf16*)carve((size_t)128 * 128 * 2);
  __bf16* nW3p = (__bf16*)carve((size_t)128 * 128 * 2);

  auto pack = [&](const float* W, __bf16* outp, int K) {
    int n = K * HD;
    pack_w_kernel<<<(n + 255) / 256, 256, 0, stream>>>(W, outp, K);
  };
  pack(mW0, mW0p, 256); pack(mW1, mW1p, 128); pack(mW2, mW2p, 128); pack(mW3, mW3p, 128);
  pack(nW0, nW0p, 384); pack(nW1, nW1p, 128); pack(nW2, nW2p, 128); pack(nW3, nW3p, 128);

  {
    int n = (int)BNH;
    embed_kernel<<<(n + 255) / 256, 256, 0, stream>>>(x, W_in, b_in, pos, xemb, h);
  }

  const int msgBlocks  = BB * GROUPS;      // 13824: one block per (batch, 3-dst group)
  const int nodeBlocks = (BB * NN) / 64;   // 648 (exact)
  const size_t msgSmem  = 4 * (16 * 256 + 16 * 128) * sizeof(__bf16)
                        + (size_t)GRP * HD * sizeof(float);             // 48 KB + 1.5 KB
  const size_t nodeSmem = 4 * (16 * 384 + 16 * 128) * sizeof(__bf16);   // 64 KB

  for (int s = 0; s < NSTEPS; ++s) {
    msg_kernel<<<msgBlocks, 128, msgSmem, stream>>>(
        h, src, dst, mW0p, mb0, mW1p, mb1, mW2p, mb2, mW3p, mb3, agg);
    node_kernel<<<nodeBlocks, 128, nodeSmem, stream>>>(
        h, xemb, agg, nW0p, nb0, nW1p, nb1, nW2p, nb2, nW3p, nb3, ln_g, ln_b);
  }
  {
    int n = BB * NN * 9;
    out_kernel<<<(n + 255) / 256, 256, 0, stream>>>(h, Wout, bout, (float*)d_out);
  }
}
